// GphSelfAttention_82154134438673
// MI455X (gfx1250) — matrix operand order repaired and checked by reading, not compiled, not run on hardware
//
#include <hip/hip_runtime.h>
#include <hip/hip_bf16.h>

// Self-attention fwd for B=2,S=2048,H=1024,NH=16,HD=64 on gfx1250 (MI455X).
// Path: fp32 -> bf16 once, all GEMMs via v_wmma_f32_16x16x32_bf16 (fp32 acc),
// flash-attention with online softmax. K/V outputs written fp32 from the
// fp32 WMMA accumulators. QKV GEMM register-blocked 32x64 per wave.

typedef __attribute__((ext_vector_type(16))) __bf16 v16bf;
typedef __attribute__((ext_vector_type(8)))  float  v8f;

#define Bq_ 2
#define Sq_ 2048
#define Hq_ 1024
#define NHq_ 16
#define HDq_ 64
#define Mq_ (Bq_ * Sq_)   // 4096 rows of hidden_states

union AFrag { v16bf v; unsigned u[8]; };
union BFrag { v16bf v; uint4 q[2]; };

// A-fragment element->K mapping (16-bit A 16x32, ISA 7.12.2):
// lane group g (0: lanes0-15, 1: lanes16-31), vgpr j holds K = base(j)+g*8+(j&3)*2 (+1)
__device__ __forceinline__ int a_koff(int j, int g) {
    return ((j < 4) ? 0 : 16) + g * 8 + (j & 3) * 2;
}

__device__ __forceinline__ float redmax16(float v) {
    #pragma unroll
    for (int m = 1; m < 16; m <<= 1) v = fmaxf(v, __shfl_xor(v, m, 32));
    return v;
}
__device__ __forceinline__ float redsum16(float v) {
    #pragma unroll
    for (int m = 1; m < 16; m <<= 1) v += __shfl_xor(v, m, 32);
    return v;
}

// ---------------- fp32 -> bf16 conversion ----------------
__global__ void cvt_bf16_kernel(const float* __restrict__ src,
                                __bf16* __restrict__ dst, int n) {
    int i = blockIdx.x * blockDim.x + threadIdx.x;
    int stride = gridDim.x * blockDim.x;
    for (; i < n; i += stride) dst[i] = (__bf16)src[i];
}

// ---------------- QKV projection GEMM ----------------
// One wave per 32x64 output tile (2 row-tiles x 1 head's 64 columns).
// which = 0:Q, 1:K, 2:V.
__global__ __launch_bounds__(256)
void qkv_gemm_kernel(const __bf16* __restrict__ hsb,   // [4096][1024] bf16
                     const __bf16* __restrict__ wb,    // [3][1024][1024] bf16
                     const float* __restrict__ bq,
                     const float* __restrict__ bk,
                     const float* __restrict__ bv,
                     float* __restrict__ out_k,        // [B,NH,S,HD] f32
                     float* __restrict__ out_v,        // [B,NH,S,HD] f32
                     __bf16* __restrict__ qb,          // [B,NH,S,HD] bf16 (scaled 1/64)
                     __bf16* __restrict__ ktb,         // [B,NH,HD,S] bf16 (transposed K)
                     __bf16* __restrict__ vbw)         // [B,NH,S,HD] bf16
{
    const int tilesH = Hq_ / 64;                       // 16 heads (column blocks)
    const int tilesM = Mq_ / 32;                       // 128 (32-row blocks)
    int wid = blockIdx.x * (blockDim.x >> 5) + (threadIdx.x >> 5);
    int which = wid / (tilesM * tilesH);
    int rem   = wid % (tilesM * tilesH);
    int tm = rem / tilesH;                             // 32-row block
    int h  = rem % tilesH;                             // head == 64-col block
    int lane = threadIdx.x & 31, g = lane >> 4, lm = lane & 15;

    const __bf16* W = wb + (size_t)which * Hq_ * Hq_;

    v8f c[2][4];
    #pragma unroll
    for (int m = 0; m < 2; ++m)
        #pragma unroll
        for (int w = 0; w < 4; ++w)
            c[m][w] = (v8f){0.f,0.f,0.f,0.f,0.f,0.f,0.f,0.f};

    const unsigned* arow0 =
        reinterpret_cast<const unsigned*>(hsb + (size_t)(tm * 32 + lm) * Hq_);
    const unsigned* arow1 =
        reinterpret_cast<const unsigned*>(hsb + (size_t)(tm * 32 + 16 + lm) * Hq_);

    for (int k0 = 0; k0 < Hq_; k0 += 32) {
        AFrag a0, a1;
        #pragma unroll
        for (int j = 0; j < 8; ++j) {
            int ko = (k0 + a_koff(j, g)) >> 1;
            a0.u[j] = arow0[ko];
            a1.u[j] = arow1[ko];
        }
        const __bf16* bp = W + (size_t)k0 * Hq_ + h * 64 + lm;   // B: lane = N col
        #pragma unroll
        for (int w = 0; w < 4; ++w) {
            BFrag b;
            #pragma unroll
            for (int s = 0; s < 16; ++s)
                b.v[s] = bp[(size_t)(a_koff(s >> 1, g) + (s & 1)) * Hq_ + w * 16];
            c[0][w] = __builtin_amdgcn_wmma_f32_16x16x32_bf16(false, a0.v, false, b.v,
                                                              (short)0, c[0][w], false, false);
            c[1][w] = __builtin_amdgcn_wmma_f32_16x16x32_bf16(false, a1.v, false, b.v,
                                                              (short)0, c[1][w], false, false);
        }
    }

    const float* bias = (which == 0) ? bq : ((which == 1) ? bk : bv);
    #pragma unroll
    for (int w = 0; w < 4; ++w) {
        int d = w * 16 + lm;                           // within-head column (C: n = lane%16)
        float bvv = bias[h * 64 + d];
        #pragma unroll
        for (int m = 0; m < 2; ++m) {
            #pragma unroll
            for (int i = 0; i < 8; ++i) {
                int row = tm * 32 + m * 16 + g * 8 + i;  // C: m = g*8 + i
                float val = c[m][w][i] + bvv;
                int bb = row >> 11, ss = row & (Sq_ - 1);
                size_t bhsd = (((size_t)(bb * NHq_ + h) * Sq_) + ss) * HDq_ + d;
                if (which == 0) {
                    qb[bhsd] = (__bf16)(val * (1.0f / 64.0f)); // fold 1/sqrt(HD) twice
                } else if (which == 1) {
                    out_k[bhsd] = val;
                    ktb[(((size_t)(bb * NHq_ + h) * HDq_) + d) * Sq_ + ss] = (__bf16)val;
                } else {
                    out_v[bhsd] = val;
                    vbw[bhsd] = (__bf16)val;
                }
            }
        }
    }
}

// ---------------- flash attention ----------------
// One wave per 16-query tile per (b,h). 32 keys per chunk, online softmax.
__global__ __launch_bounds__(256)
void attn_kernel(const __bf16* __restrict__ qb,   // [B,NH,S,HD] scaled
                 const __bf16* __restrict__ ktb,  // [B,NH,HD,S]
                 const __bf16* __restrict__ vbw,  // [B,NH,S,HD]
                 float* __restrict__ ctx)         // [B,S,H]
{
    __shared__ __bf16 psh[8 * 16 * 32];            // 1KB per wave P-transpose tile
    int wslot = threadIdx.x >> 5;
    int wid = blockIdx.x * 8 + wslot;
    const int qtiles = Sq_ / 16;                   // 128
    int b  = wid / (NHq_ * qtiles);
    int rm = wid % (NHq_ * qtiles);
    int h  = rm / qtiles;
    int qt = rm % qtiles;
    int lane = threadIdx.x & 31, g = lane >> 4, lm = lane & 15;

    size_t bh = (size_t)(b * NHq_ + h);

    // Preload Q A-fragments (16 rows x 64 head-dim = 2 fragments)
    AFrag qa[2];
    {
        const unsigned* qrow =
            reinterpret_cast<const unsigned*>(qb + (bh * Sq_ + (size_t)qt * 16 + lm) * HDq_);
        #pragma unroll
        for (int f = 0; f < 2; ++f)
            #pragma unroll
            for (int j = 0; j < 8; ++j)
                qa[f].u[j] = qrow[(f * 32 + a_koff(j, g)) >> 1];
    }

    v8f o[4];
    #pragma unroll
    for (int w = 0; w < 4; ++w) o[w] = (v8f){0.f,0.f,0.f,0.f,0.f,0.f,0.f,0.f};
    float mi[8], li[8];
    #pragma unroll
    for (int i = 0; i < 8; ++i) { mi[i] = -3.0e38f; li[i] = 0.f; }

    __bf16* myp = psh + wslot * 512;

    for (int chunk = 0; chunk < Sq_ / 32; ++chunk) {
        int key0 = chunk * 32;

        // ---- scores: S(16x32) = Q(16x64) @ K_chunk^T ----
        v8f s[2];
        #pragma unroll
        for (int t = 0; t < 2; ++t) {
            BFrag kb0, kb1;                        // d in [0,32) and [32,64)
            {
                const __bf16* kp = ktb + bh * HDq_ * Sq_ + key0 + t * 16 + lm;
                #pragma unroll
                for (int s = 0; s < 16; ++s) {
                    int dk = a_koff(s >> 1, g) + (s & 1);
                    kb0.v[s] = kp[(size_t)dk * Sq_];
                    kb1.v[s] = kp[(size_t)(32 + dk) * Sq_];
                }
            }
            v8f z = {0.f,0.f,0.f,0.f,0.f,0.f,0.f,0.f};
            z = __builtin_amdgcn_wmma_f32_16x16x32_bf16(false, qa[0].v, false, kb0.v,
                                                        (short)0, z, false, false);
            s[t] = __builtin_amdgcn_wmma_f32_16x16x32_bf16(false, qa[1].v, false, kb1.v,
                                                           (short)0, z, false, false);
        }

        // ---- online softmax (rows live across 16 lanes within a half-wave) ----
        float alpha[8];
        #pragma unroll
        for (int i = 0; i < 8; ++i) {
            float rmax = redmax16(fmaxf(s[0][i], s[1][i]));
            float mn = fmaxf(mi[i], rmax);
            float al = __expf(mi[i] - mn);
            float p0 = __expf(s[0][i] - mn);
            float p1 = __expf(s[1][i] - mn);
            float rs = redsum16(p0 + p1);
            li[i] = li[i] * al + rs;
            mi[i] = mn;
            alpha[i] = al;
            s[0][i] = p0;
            s[1][i] = p1;
        }
        #pragma unroll
        for (int w = 0; w < 4; ++w)
            #pragma unroll
            for (int i = 0; i < 8; ++i) o[w][i] *= alpha[i];

        // ---- transpose P (C-layout) -> A-layout via LDS ----
        #pragma unroll
        for (int t = 0; t < 2; ++t)
            #pragma unroll
            for (int i = 0; i < 8; ++i)
                myp[(g * 8 + i) * 32 + t * 16 + lm] = (__bf16)s[t][i];
        asm volatile("s_wait_dscnt 0" ::: "memory");
        __builtin_amdgcn_wave_barrier();

        AFrag pa;
        const unsigned* pr = reinterpret_cast<const unsigned*>(myp);
        #pragma unroll
        for (int j = 0; j < 8; ++j)
            pa.u[j] = pr[(lm * 32 + a_koff(j, g)) >> 1];
        __builtin_amdgcn_wave_barrier();           // DS ops are in-order per wave (WAR safe)

        // ---- O(16x64) += P(16x32) @ V_chunk(32x64) ----
        const __bf16* vrow = vbw + (bh * Sq_ + key0) * HDq_ + lm;   // B: lane = N col
        #pragma unroll
        for (int w = 0; w < 4; ++w) {
            BFrag vf;
            #pragma unroll
            for (int s = 0; s < 16; ++s)
                vf.v[s] = vrow[(size_t)(a_koff(s >> 1, g) + (s & 1)) * HDq_ + w * 16];
            o[w] = __builtin_amdgcn_wmma_f32_16x16x32_bf16(false, pa.v, false, vf.v,
                                                           (short)0, o[w], false, false);
        }
    }

    // ---- epilogue: divide by row sums, write ctx [B,S,H] ----
    float inv[8];
    #pragma unroll
    for (int i = 0; i < 8; ++i) inv[i] = 1.0f / li[i];
    #pragma unroll
    for (int w = 0; w < 4; ++w)
        #pragma unroll
        for (int i = 0; i < 8; ++i) {
            int srow = qt * 16 + g * 8 + i;
            int col = h * 64 + w * 16 + lm;
            ctx[((size_t)b * Sq_ + srow) * Hq_ + col] = o[w][i] * inv[i];
        }
}

extern "C" void kernel_launch(void* const* d_in, const int* in_sizes, int n_in,
                              void* d_out, int out_size, void* d_ws, size_t ws_size,
                              hipStream_t stream) {
    const float* hs = (const float*)d_in[0];
    const float* Wq = (const float*)d_in[1];
    const float* bq = (const float*)d_in[2];
    const float* Wk = (const float*)d_in[3];
    const float* bk = (const float*)d_in[4];
    const float* Wv = (const float*)d_in[5];
    const float* bv = (const float*)d_in[6];

    float* out_ctx = (float*)d_out;                         // [B,S,H]
    float* out_k = out_ctx + (size_t)Bq_ * Sq_ * Hq_;       // [B,NH,S,HD]
    float* out_v = out_k + (size_t)Bq_ * Sq_ * Hq_;         // [B,NH,S,HD]

    // Workspace carve (bf16): hsb 8MiB | Wb 6MiB | qb 8MiB | ktb 8MiB | vb 8MiB
    __bf16* hsb = (__bf16*)d_ws;
    __bf16* wb  = hsb + (size_t)Mq_ * Hq_;
    __bf16* qb  = wb + (size_t)3 * Hq_ * Hq_;
    __bf16* ktb = qb + (size_t)Mq_ * Hq_;
    __bf16* vbw = ktb + (size_t)Mq_ * Hq_;

    // 1) fp32 -> bf16 conversions
    cvt_bf16_kernel<<<2048, 256, 0, stream>>>(hs, hsb, Mq_ * Hq_);
    cvt_bf16_kernel<<<1024, 256, 0, stream>>>(Wq, wb + 0 * (size_t)Hq_ * Hq_, Hq_ * Hq_);
    cvt_bf16_kernel<<<1024, 256, 0, stream>>>(Wk, wb + 1 * (size_t)Hq_ * Hq_, Hq_ * Hq_);
    cvt_bf16_kernel<<<1024, 256, 0, stream>>>(Wv, wb + 2 * (size_t)Hq_ * Hq_, Hq_ * Hq_);

    // 2) QKV projections: 3 * 128 * 16 = 6144 wave-tiles (32x64 each), 8 waves/block
    qkv_gemm_kernel<<<768, 256, 0, stream>>>(hsb, wb, bq, bk, bv,
                                             out_k, out_v, qb, ktb, vbw);

    // 3) attention: 2*16*128 = 4096 wave-tiles, 8 waves/block
    attn_kernel<<<512, 256, 0, stream>>>(qb, ktb, vbw, out_ctx);
}
